// TDrumorGAT_25494925869740
// MI455X (gfx1250) — matrix operand order repaired and checked by reading, not compiled, not run on hardware
//
#include <hip/hip_runtime.h>

// ---------------------------------------------------------------------------
// TDrumorGAT forward for gfx1250 (MI455X).
// GEMMs: v_wmma_f32_16x16x32_bf16 with
//   - weights pre-transposed/pre-converted to bf16 (contiguous b128 B loads)
//   - A tiles staged through LDS once per block (ds_load_b128 fragments)
// Edge softmax/aggregation: encoded-u32 atomicMax + f32 atomicAdd (L2-resident).
// ---------------------------------------------------------------------------

#define NNODES 50000
#define NEDGES 400000
#define NGRAPH 128
#define FIN 768
#define HIDF 256
#define FOUT 128
#define NHEAD 4
#define NEG_SLOPE 0.2f
#define EN (NEDGES + NNODES)
#define POOLD (FOUT + HIDF) /* 384 */
#define KCAT (HIDF + FIN)   /* 1024 */

typedef __bf16 bf16_t;
typedef bf16_t v16bf __attribute__((ext_vector_type(16)));
typedef bf16_t v8bf __attribute__((ext_vector_type(8)));
typedef bf16_t v2bf __attribute__((ext_vector_type(2)));
typedef float v8f __attribute__((ext_vector_type(8)));

__device__ __forceinline__ bf16_t tobf(float f) { return (bf16_t)f; }

__device__ __forceinline__ v8f wmma_bf16(v16bf a, v16bf b, v8f c) {
  // 8 args: (neg_a, A, neg_b, B, c_mod, C, reuse_a, reuse_b)
  return __builtin_amdgcn_wmma_f32_16x16x32_bf16(false, a, false, b,
                                                 (short)0, c, false, false);
}
__device__ __forceinline__ v16bf cat8(v8bf lo, v8bf hi) {
  return __builtin_shufflevector(lo, hi, 0, 1, 2, 3, 4, 5, 6, 7, 8, 9, 10, 11,
                                 12, 13, 14, 15);
}

// Monotone float<->uint encoding so atomicMax(u32) implements float max.
__device__ __forceinline__ unsigned fenc(float f) {
  unsigned u = __float_as_uint(f);
  return (u & 0x80000000u) ? ~u : (u | 0x80000000u);
}
__device__ __forceinline__ float fdec(unsigned e) {
  unsigned u = (e & 0x80000000u) ? (e ^ 0x80000000u) : ~e;
  return __uint_as_float(u);
}

// ---------------------------------------------------------------------------
// Weight prep: Wt[c][k] = bf16(W[k][c])   (W is [K, C] row-major f32)
// ---------------------------------------------------------------------------
__global__ void transpose_bf16_k(const float* __restrict__ W,
                                 bf16_t* __restrict__ Wt, int K, int C) {
  int t = blockIdx.x * blockDim.x + threadIdx.x;
  if (t >= K * C) return;
  int c = t / K, k = t - c * K;
  Wt[(size_t)c * K + k] = tobf(W[(size_t)k * C + c]);
}

// ---------------------------------------------------------------------------
// GEMM1: Hlin1[N,256] = x[N,768] @ W1 ; B from W1t bf16 [256][768].
// Block = 256 threads = 8 waves; block -> 16 rows, wave w -> cols [32w,32w+32).
// ---------------------------------------------------------------------------
__global__ __launch_bounds__(256) void gemm_x_w1(
    const float* __restrict__ x, const bf16_t* __restrict__ W1t,
    float* __restrict__ out) {
  __shared__ __align__(16) bf16_t Atile[16 * 32];  // 16 rows x 32 K, bf16
  const int tid = threadIdx.x;
  const int lane = tid & 31;
  const int wave = tid >> 5;
  const bool lo = lane < 16;
  const int m = lane & 15;
  const int akb = lo ? 0 : 8;   // A K-base per lane half (16x32 bf16 layout)
  const int bkb = lo ? 0 : 8;   // B K-base per lane half (32x16 bf16 layout)
  const int col0 = wave * 32 + m;
  const int col1 = col0 + 16;

  // staging assignment: thread -> 2 consecutive K of one row
  const int srow = (tid * 2) >> 5;
  const int skcol = (tid * 2) & 31;
  const float* sp = x + (size_t)(blockIdx.x * 16 + srow) * FIN + skcol;

  const bf16_t* b0p = W1t + (size_t)col0 * FIN + bkb;
  const bf16_t* b1p = W1t + (size_t)col1 * FIN + bkb;
  bf16_t* stp = Atile + tid * 2;
  const bf16_t* arp = Atile + m * 32 + akb;

  v8f acc0 = {};
  v8f acc1 = {};
  for (int k0 = 0; k0 < FIN; k0 += 32) {
    __syncthreads();  // previous iteration's fragment reads complete
    float2 f = *(const float2*)(sp + k0);
    if (k0 + 32 < FIN) __builtin_prefetch(sp + k0 + 32, 0, 3);
    v2bf p;
    p[0] = tobf(f.x);
    p[1] = tobf(f.y);
    *(v2bf*)stp = p;
    __syncthreads();  // tile staged

    v8bf alo = *(const v8bf*)(arp);       // K = akb .. akb+7
    v8bf ahi = *(const v8bf*)(arp + 16);  // K = akb+16 .. akb+23
    v16bf a = cat8(alo, ahi);
    v16bf b0 = cat8(*(const v8bf*)(b0p + k0), *(const v8bf*)(b0p + k0 + 16));
    v16bf b1 = cat8(*(const v8bf*)(b1p + k0), *(const v8bf*)(b1p + k0 + 16));
    acc0 = wmma_bf16(a, b0, acc0);
    acc1 = wmma_bf16(a, b1, acc1);
  }

  const int rbase = blockIdx.x * 16 + (lo ? 0 : 8);
#pragma unroll
  for (int r = 0; r < 8; ++r) {
    out[(size_t)(rbase + r) * HIDF + col0] = acc0[r];
    out[(size_t)(rbase + r) * HIDF + col1] = acc1[r];
  }
}

// ---------------------------------------------------------------------------
// GEMM2: Hlin2[N,128] = relu(concat(h1[N,256], x[root[n],:])) @ W2
// Concat+relu applied during LDS staging; B from W2t bf16 [128][1024].
// Block = 256 threads = 8 waves; wave w -> cols [16w,16w+16).
// ---------------------------------------------------------------------------
__global__ __launch_bounds__(256) void gemm_hcat_w2(
    const float* __restrict__ h1, const float* __restrict__ x,
    const int* __restrict__ rootidx, const bf16_t* __restrict__ W2t,
    float* __restrict__ out) {
  __shared__ __align__(16) bf16_t Atile[16 * 32];
  const int tid = threadIdx.x;
  const int lane = tid & 31;
  const int wave = tid >> 5;
  const bool lo = lane < 16;
  const int m = lane & 15;
  const int akb = lo ? 0 : 8;
  const int bkb = lo ? 0 : 8;
  const int col = wave * 16 + m;

  const int srow = (tid * 2) >> 5;
  const int skcol = (tid * 2) & 31;
  const int n = blockIdx.x * 16 + srow;
  const float* h1s = h1 + (size_t)n * HIDF + skcol;
  const float* xs = x + (size_t)rootidx[n] * FIN + skcol;

  const bf16_t* bp = W2t + (size_t)col * KCAT + bkb;
  bf16_t* stp = Atile + tid * 2;
  const bf16_t* arp = Atile + m * 32 + akb;

  v8f acc = {};
  for (int k0 = 0; k0 < KCAT; k0 += 32) {
    __syncthreads();
    const float* p = (k0 < HIDF) ? (h1s + k0) : (xs + (k0 - HIDF));
    float2 f = *(const float2*)p;
    v2bf pk;
    pk[0] = tobf(fmaxf(f.x, 0.f));
    pk[1] = tobf(fmaxf(f.y, 0.f));
    *(v2bf*)stp = pk;
    __syncthreads();

    v16bf a = cat8(*(const v8bf*)(arp), *(const v8bf*)(arp + 16));
    v16bf b = cat8(*(const v8bf*)(bp + k0), *(const v8bf*)(bp + k0 + 16));
    acc = wmma_bf16(a, b, acc);
  }

  const int rbase = blockIdx.x * 16 + (lo ? 0 : 8);
#pragma unroll
  for (int r = 0; r < 8; ++r)
    out[(size_t)(rbase + r) * FOUT + col] = acc[r];
}

// ---------------------------------------------------------------------------
// Per-node attention logits: asb/adb[n,h] = sum_c Hlin[n,h,c] * a_{s,d}[h,c]
// ---------------------------------------------------------------------------
__global__ void alphas_k(const float* __restrict__ Hlin,
                         const float* __restrict__ a_s,
                         const float* __restrict__ a_d,
                         float* __restrict__ asb, float* __restrict__ adb,
                         int C) {
  int t = blockIdx.x * blockDim.x + threadIdx.x;
  if (t >= NNODES * NHEAD) return;
  int n = t >> 2, h = t & 3;
  const float* hp = Hlin + (size_t)n * (NHEAD * C) + h * C;
  const float* sp = a_s + h * C;
  const float* dp = a_d + h * C;
  float s = 0.f, d = 0.f;
  for (int c = 0; c < C; ++c) {
    float v = hp[c];
    s += v * sp[c];
    d += v * dp[c];
  }
  asb[t] = s;
  adb[t] = d;
}

__device__ __forceinline__ void edge_sd(int e, const int* src, const int* dst,
                                        int& s, int& d) {
  if (e < NEDGES) { s = src[e]; d = dst[e]; }
  else { s = e - NEDGES; d = e - NEDGES; }  // self-loops
}

__global__ void edge_max_k(const int* __restrict__ src,
                           const int* __restrict__ dst,
                           const float* __restrict__ asb,
                           const float* __restrict__ adb,
                           unsigned* __restrict__ maxb) {
  int t = blockIdx.x * blockDim.x + threadIdx.x;
  if (t >= EN * NHEAD) return;
  int e = t >> 2, h = t & 3, s, d;
  edge_sd(e, src, dst, s, d);
  float v = asb[s * NHEAD + h] + adb[d * NHEAD + h];
  v = v > 0.f ? v : NEG_SLOPE * v;
  atomicMax(maxb + d * NHEAD + h, fenc(v));
}

__global__ void edge_expw_k(const int* __restrict__ src,
                            const int* __restrict__ dst,
                            const float* __restrict__ asb,
                            const float* __restrict__ adb,
                            const unsigned* __restrict__ maxb,
                            float* __restrict__ wbuf,
                            float* __restrict__ denom) {
  int t = blockIdx.x * blockDim.x + threadIdx.x;
  if (t >= EN * NHEAD) return;
  int e = t >> 2, h = t & 3, s, d;
  edge_sd(e, src, dst, s, d);
  float v = asb[s * NHEAD + h] + adb[d * NHEAD + h];
  v = v > 0.f ? v : NEG_SLOPE * v;
  float w = __expf(v - fdec(maxb[d * NHEAD + h]));
  wbuf[t] = w;
  atomicAdd(denom + d * NHEAD + h, w);
}

__global__ void edge_agg_k(const int* __restrict__ src,
                           const int* __restrict__ dst,
                           const float* __restrict__ wbuf,
                           const float* __restrict__ denom,
                           const float* __restrict__ Hlin,
                           float* __restrict__ Hout, int C) {
  int t = blockIdx.x * blockDim.x + threadIdx.x;
  if (t >= EN * NHEAD) return;
  int e = t >> 2, h = t & 3, s, d;
  edge_sd(e, src, dst, s, d);
  float alpha = wbuf[t] / denom[d * NHEAD + h];
  const float* hp = Hlin + (size_t)s * (NHEAD * C) + h * C;
  float* op = Hout + (size_t)d * (NHEAD * C) + h * C;
  for (int c = 0; c < C; ++c) atomicAdd(op + c, hp[c] * alpha);
}

// root pointers: batch is sorted; first node of each graph is its root
__global__ void root_start_k(const int* __restrict__ batch,
                             int* __restrict__ rootptr) {
  int n = blockIdx.x * blockDim.x + threadIdx.x;
  if (n >= NNODES) return;
  if (n == 0 || batch[n] != batch[n - 1]) rootptr[batch[n]] = n;
}
__global__ void root_gather_k(const int* __restrict__ batch,
                              const int* __restrict__ rootptr,
                              int* __restrict__ rootidx) {
  int n = blockIdx.x * blockDim.x + threadIdx.x;
  if (n >= NNODES) return;
  rootidx[n] = rootptr[batch[n]];
}

__global__ void bias_add_k(float* __restrict__ H, const float* __restrict__ b,
                           int total, int D) {
  int t = blockIdx.x * blockDim.x + threadIdx.x;
  if (t < total) H[t] += b[t % D];
}
__global__ void bias_relu_k(float* __restrict__ H, const float* __restrict__ b,
                            int total, int D) {
  int t = blockIdx.x * blockDim.x + threadIdx.x;
  if (t < total) H[t] = fmaxf(H[t] + b[t % D], 0.f);
}

__global__ void fill_f32_k(float* __restrict__ p, float v, int n) {
  int t = blockIdx.x * blockDim.x + threadIdx.x;
  if (t < n) p[t] = v;
}
__global__ void fill_u32_k(unsigned* __restrict__ p, unsigned v, int n) {
  int t = blockIdx.x * blockDim.x + threadIdx.x;
  if (t < n) p[t] = v;
}

__global__ void pool_count_k(const int* __restrict__ batch,
                             float* __restrict__ counts) {
  int n = blockIdx.x * blockDim.x + threadIdx.x;
  if (n >= NNODES) return;
  atomicAdd(counts + batch[n], 1.f);
}
__global__ void pool_sum_k(const int* __restrict__ batch,
                           const int* __restrict__ rootidx,
                           const float* __restrict__ Hout2,
                           const float* __restrict__ Hout1,
                           float* __restrict__ out) {
  int t = blockIdx.x * blockDim.x + threadIdx.x;
  if (t >= NNODES * POOLD) return;
  int n = t / POOLD, k = t - n * POOLD;
  float v = (k < FOUT) ? Hout2[(size_t)n * FOUT + k]
                       : Hout1[(size_t)rootidx[n] * HIDF + (k - FOUT)];
  atomicAdd(out + (size_t)batch[n] * POOLD + k, v);
}
__global__ void pool_div_k(float* __restrict__ out,
                           const float* __restrict__ counts) {
  int t = blockIdx.x * blockDim.x + threadIdx.x;
  if (t >= NGRAPH * POOLD) return;
  out[t] /= fmaxf(counts[t / POOLD], 1.f);
}

// ---------------------------------------------------------------------------
extern "C" void kernel_launch(void* const* d_in, const int* in_sizes, int n_in,
                              void* d_out, int out_size, void* d_ws,
                              size_t ws_size, hipStream_t stream) {
  const float* x = (const float*)d_in[0];
  const int* src = (const int*)d_in[1];
  const int* dst = (const int*)d_in[2];
  const int* batch = (const int*)d_in[3];
  const float* W1 = (const float*)d_in[4];
  const float* as1 = (const float*)d_in[5];
  const float* ad1 = (const float*)d_in[6];
  const float* b1 = (const float*)d_in[7];
  const float* W2 = (const float*)d_in[8];
  const float* as2 = (const float*)d_in[9];
  const float* ad2 = (const float*)d_in[10];
  const float* b2 = (const float*)d_in[11];
  float* out = (float*)d_out;

  // deterministic workspace carve-out (256B aligned)
  char* ws = (char*)d_ws;
  size_t off = 0;
  auto carve = [&](size_t bytes) {
    char* p = ws + off;
    off = (off + bytes + 255) & ~(size_t)255;
    return p;
  };
  float* Hlin1 = (float*)carve((size_t)NNODES * HIDF * 4);
  float* Hout1 = (float*)carve((size_t)NNODES * HIDF * 4);
  float* Hlin2 = (float*)carve((size_t)NNODES * FOUT * 4);
  float* Hout2 = (float*)carve((size_t)NNODES * FOUT * 4);
  float* asb = (float*)carve((size_t)NNODES * NHEAD * 4);
  float* adb = (float*)carve((size_t)NNODES * NHEAD * 4);
  unsigned* maxb = (unsigned*)carve((size_t)NNODES * NHEAD * 4);
  float* denom = (float*)carve((size_t)NNODES * NHEAD * 4);
  float* wbuf = (float*)carve((size_t)EN * NHEAD * 4);
  int* rootptr = (int*)carve((size_t)NGRAPH * 4);
  int* rootidx = (int*)carve((size_t)NNODES * 4);
  float* counts = (float*)carve((size_t)NGRAPH * 4);
  bf16_t* W1t = (bf16_t*)carve((size_t)FIN * HIDF * 2);
  bf16_t* W2t = (bf16_t*)carve((size_t)KCAT * FOUT * 2);
  (void)ws_size; (void)n_in; (void)in_sizes; (void)out_size;

  const unsigned NEGINF_ENC = 0x007FFFFFu;  // fenc(-inf)
  auto G = [](int n) { return dim3((unsigned)((n + 255) / 256)); };
  const dim3 B(256);

  // ---- weight prep (bf16, transposed to [col][K]) ----
  transpose_bf16_k<<<G(FIN * HIDF), B, 0, stream>>>(W1, W1t, FIN, HIDF);
  transpose_bf16_k<<<G(KCAT * FOUT), B, 0, stream>>>(W2, W2t, KCAT, FOUT);

  // ---- layer 1 ----
  fill_f32_k<<<G(NNODES * HIDF), B, 0, stream>>>(Hout1, 0.f, NNODES * HIDF);
  fill_f32_k<<<G(NNODES * NHEAD), B, 0, stream>>>(denom, 0.f, NNODES * NHEAD);
  fill_u32_k<<<G(NNODES * NHEAD), B, 0, stream>>>(maxb, NEGINF_ENC,
                                                  NNODES * NHEAD);
  gemm_x_w1<<<dim3(NNODES / 16), B, 0, stream>>>(x, W1t, Hlin1);
  alphas_k<<<G(NNODES * NHEAD), B, 0, stream>>>(Hlin1, as1, ad1, asb, adb, 64);
  root_start_k<<<G(NNODES), B, 0, stream>>>(batch, rootptr);
  root_gather_k<<<G(NNODES), B, 0, stream>>>(batch, rootptr, rootidx);
  edge_max_k<<<G(EN * NHEAD), B, 0, stream>>>(src, dst, asb, adb, maxb);
  edge_expw_k<<<G(EN * NHEAD), B, 0, stream>>>(src, dst, asb, adb, maxb, wbuf,
                                               denom);
  edge_agg_k<<<G(EN * NHEAD), B, 0, stream>>>(src, dst, wbuf, denom, Hlin1,
                                              Hout1, 64);
  bias_add_k<<<G(NNODES * HIDF), B, 0, stream>>>(Hout1, b1, NNODES * HIDF,
                                                 HIDF);

  // ---- layer 2 ----
  fill_f32_k<<<G(NNODES * FOUT), B, 0, stream>>>(Hout2, 0.f, NNODES * FOUT);
  fill_f32_k<<<G(NNODES * NHEAD), B, 0, stream>>>(denom, 0.f, NNODES * NHEAD);
  fill_u32_k<<<G(NNODES * NHEAD), B, 0, stream>>>(maxb, NEGINF_ENC,
                                                  NNODES * NHEAD);
  gemm_hcat_w2<<<dim3(NNODES / 16), B, 0, stream>>>(Hout1, x, rootidx, W2t,
                                                    Hlin2);
  alphas_k<<<G(NNODES * NHEAD), B, 0, stream>>>(Hlin2, as2, ad2, asb, adb, 32);
  edge_max_k<<<G(EN * NHEAD), B, 0, stream>>>(src, dst, asb, adb, maxb);
  edge_expw_k<<<G(EN * NHEAD), B, 0, stream>>>(src, dst, asb, adb, maxb, wbuf,
                                               denom);
  edge_agg_k<<<G(EN * NHEAD), B, 0, stream>>>(src, dst, wbuf, denom, Hlin2,
                                              Hout2, 32);
  bias_relu_k<<<G(NNODES * FOUT), B, 0, stream>>>(Hout2, b2, NNODES * FOUT,
                                                  FOUT);

  // ---- pooling: scatter_mean over graphs ----
  fill_f32_k<<<G(NGRAPH * POOLD), B, 0, stream>>>(out, 0.f, NGRAPH * POOLD);
  fill_f32_k<<<G(NGRAPH), B, 0, stream>>>(counts, 0.f, NGRAPH);
  pool_count_k<<<G(NNODES), B, 0, stream>>>(batch, counts);
  pool_sum_k<<<G(NNODES * POOLD), B, 0, stream>>>(batch, rootidx, Hout2, Hout1,
                                                  out);
  pool_div_k<<<G(NGRAPH * POOLD), B, 0, stream>>>(out, counts);
}
